// EdgeRegressionModel_14508399526310
// MI455X (gfx1250) — matrix operand order repaired and checked by reading, compiled, not run on hardware
//
#include <hip/hip_runtime.h>

// ---------------------------------------------------------------------------
// EdgeRegression GNN, MI455X (gfx1250) implementation.
// All GEMMs use v_wmma_f32_16x16x32_bf16 (wave32, 16x16 tiles, f32 accum).
// Edge activations stored bf16 (HBM-bound workload), node tensors L2-resident.
// BN folded into per-column scale/shift applied on-the-fly in next GEMM.
// Output tiles staged through per-wave LDS slices -> one b128 store per lane.
// ---------------------------------------------------------------------------

typedef unsigned short u16;
typedef unsigned int   u32;
typedef __attribute__((ext_vector_type(16))) __bf16 v16bf;
typedef __attribute__((ext_vector_type(8)))  float  v8f;

#define N_NODES 50000
#define N_EDGES 800000
#define NLAYERS 3
#define BN_EPS  1e-5f

// Workspace layout (bytes)
#define OFF_WLIN   ((size_t)0)          //  4096  : lin_in packed (K=32)
#define OFF_WMSG1  ((size_t)4096)       // 73728  : msg_w1 packed (3 x K=192)
#define OFF_WMSG2  ((size_t)77824)      // 24576  : msg_w2 packed (3 x K=64)
#define OFF_WUPD1  ((size_t)102400)     // 49152  : upd_w1 packed (3 x K=128)
#define OFF_WUPD2  ((size_t)151552)     // 24576  : upd_w2 packed (3 x K=64)
#define OFF_STATS  ((size_t)176128)     // 12288  : 12 stages x [sum|ss|scale|shift] x 64 f32
#define OFF_H      ((size_t)188416)     // 12.8MB : h f32 [N,64]
#define OFF_HBF    ((size_t)12988416)   //  6.4MB : h bf16
#define OFF_AGG    ((size_t)19388416)   // 12.8MB : agg f32 [N,64]
#define OFF_UBF    ((size_t)32188416)   //  6.4MB : u pre-BN bf16 [N,64] (in-place GEMM2)
#define OFF_E      ((size_t)38588416)   //204.8MB : e f32 [E,64]
#define OFF_EBF    ((size_t)243388416)  //102.4MB : e bf16
#define OFF_MBF    ((size_t)345788416)  //102.4MB : m pre-BN bf16 [E,64] (in-place GEMM2)

union FragBF {
  v16bf v;
  uint4 q[2];
  u16   s[16];
};

static __device__ __forceinline__ float bf2f(u16 h) {
  union { u32 u; float f; } c; c.u = ((u32)h) << 16; return c.f;
}
static __device__ __forceinline__ u16 f2bf(float f) {
  union { float f; u32 u; } c; c.f = f;
  u32 u = c.u;
  u32 r = u + 0x7FFFu + ((u >> 16) & 1u);   // round-to-nearest-even
  return (u16)(r >> 16);
}
static __device__ __forceinline__ u32 pack2bf(float a, float b) {
  return (u32)f2bf(a) | ((u32)f2bf(b) << 16);
}

// Stage a 16x16 f32 accumulator tile through this wave's LDS slice and emit
// coalesced bf16 stores: each lane writes one 16-byte chunk (half an output
// row). LDS ops within a wave are in-order (DScnt), so no barrier is needed
// for the intra-wave transpose. Stats (masked) accumulated from registers.
template <bool GUARD>
static __device__ __forceinline__ void tile_writeout(
    float (*st)[17], const v8f& c, int lane, int half, int m,
    u16* __restrict__ out, int row0, int n0, int rows,
    float& s, float& ss) {
#pragma unroll
  for (int r = 0; r < 8; ++r) {
    int   tr = r + half * 8;
    float y  = c[r];
    st[tr][m] = y;
    float w = (!GUARD || (row0 + tr) < rows) ? 1.f : 0.f;
    s  += w * y;
    ss += w * y * y;
  }
  int tr2 = lane >> 1;     // output row within tile
  int seg = lane & 1;      // which 8-col half of the row
  uint4 q;
  q.x = pack2bf(st[tr2][seg * 8 + 0], st[tr2][seg * 8 + 1]);
  q.y = pack2bf(st[tr2][seg * 8 + 2], st[tr2][seg * 8 + 3]);
  q.z = pack2bf(st[tr2][seg * 8 + 4], st[tr2][seg * 8 + 5]);
  q.w = pack2bf(st[tr2][seg * 8 + 6], st[tr2][seg * 8 + 7]);
  int grow = row0 + tr2;
  if (!GUARD || grow < rows)
    *(uint4*)(out + (size_t)grow * 64 + n0 + seg * 8) = q;
}

// ---------------------------------------------------------------------------
__global__ void k_zero(float* __restrict__ p, int n) {
  int i = blockIdx.x * 256 + threadIdx.x;
  if (i < n) p[i] = 0.f;
}

// Pack a [K x 64] f32 row-major weight into WMMA B-fragment order, bf16.
// Layout: dst[((chunk*4 + ntile)*32 + lane)*16 + j]
//   lane half = lane>>4, row = chunk*32 + (j<8 ? half*8+j : 16+half*8+(j-8)), col = ntile*16 + (lane&15)
__global__ void k_pack_w(const float* __restrict__ src, u16* __restrict__ dst, int K) {
  int idx = blockIdx.x * 256 + threadIdx.x;
  int total = (K >> 5) * 2048;
  if (idx >= total) return;
  int j    = idx & 15;
  int lane = (idx >> 4) & 31;
  int nt   = (idx >> 9) & 3;
  int ch   = idx >> 11;
  int half = lane >> 4;
  int row  = ch * 32 + ((j < 8) ? (half * 8 + j) : (16 + half * 8 + (j - 8)));
  int col  = nt * 16 + (lane & 15);
  dst[idx] = f2bf(src[row * 64 + col]);
}

// h = x @ lin_in_w + lin_in_b : [N,32] x [32,64], single K-chunk WMMA
__global__ void k_h_init(const float* __restrict__ x, const u16* __restrict__ wpack,
                         const float* __restrict__ bias,
                         float* __restrict__ h, u16* __restrict__ hbf) {
  int tid  = threadIdx.x;
  int lane = tid & 31, wave = tid >> 5;
  int half = lane >> 4, m = lane & 15;
  int nt   = wave & 3,  rowg = wave >> 2;
  int n0   = nt * 16;
  int row0 = blockIdx.x * 32 + rowg * 16;
  int ar   = row0 + m;
  FragBF A;
  if (ar < N_NODES) {
    const float* xr = x + (size_t)ar * 32;
#pragma unroll
    for (int j = 0; j < 8; ++j) A.s[j]     = f2bf(xr[half * 8 + j]);
#pragma unroll
    for (int j = 0; j < 8; ++j) A.s[8 + j] = f2bf(xr[16 + half * 8 + j]);
  } else {
    A.q[0] = make_uint4(0, 0, 0, 0);
    A.q[1] = make_uint4(0, 0, 0, 0);
  }
  FragBF B;
  const u16* wp = wpack + ((size_t)(nt * 32 + lane)) * 16;
  B.q[0] = *(const uint4*)wp;
  B.q[1] = *(const uint4*)(wp + 8);
  v8f c = {};
  c = __builtin_amdgcn_wmma_f32_16x16x32_bf16(false, A.v, false, B.v, (short)0, c, false, false);
  int col = n0 + m;
  float bv = bias[col];
#pragma unroll
  for (int r = 0; r < 8; ++r) {
    int gr = row0 + r + half * 8;
    if (gr < N_NODES) {
      float y = c[r] + bv;
      h[(size_t)gr * 64 + col]   = y;
      hbf[(size_t)gr * 64 + col] = f2bf(y);
    }
  }
}

// e = attr[:,None] @ edg_in_w + edg_in_b (rank-1), stored f32 + bf16
__global__ void k_e_init(const float* __restrict__ attr, const float* __restrict__ ew,
                         const float* __restrict__ eb,
                         float* __restrict__ e32, u16* __restrict__ ebf) {
  int t = blockIdx.x * 256 + threadIdx.x;
  if (t >= N_EDGES * 16) return;
  int e = t >> 4, c0 = (t & 15) * 4;
  float a = attr[e];
  float y0 = fmaf(a, ew[c0 + 0], eb[c0 + 0]);
  float y1 = fmaf(a, ew[c0 + 1], eb[c0 + 1]);
  float y2 = fmaf(a, ew[c0 + 2], eb[c0 + 2]);
  float y3 = fmaf(a, ew[c0 + 3], eb[c0 + 3]);
  float4 v; v.x = y0; v.y = y1; v.z = y2; v.w = y3;
  *(float4*)(e32 + (size_t)e * 64 + c0) = v;
  uint2 pk; pk.x = pack2bf(y0, y1); pk.y = pack2bf(y2, y3);
  *(uint2*)(ebf + (size_t)e * 64 + c0) = pk;
}

// msg GEMM1: concat(h[dst], h[src], e) [E,192] @ W1[192,64] -> mbf pre-BN + stats
__global__ void k_msg_gemm1(const int* __restrict__ eidx, const u16* __restrict__ hbf,
                            const u16* __restrict__ ebf, const u16* __restrict__ wpack,
                            u16* __restrict__ mbf,
                            float* __restrict__ gsum, float* __restrict__ gss) {
  __shared__ float lsum[64], lss[64];
  __shared__ float stile[8][16][17];
  int tid = threadIdx.x;
  if (tid < 64) { lsum[tid] = 0.f; lss[tid] = 0.f; }
  int lane = tid & 31, wave = tid >> 5, half = lane >> 4, m = lane & 15;
  int nt = wave & 3, rowg = wave >> 2, n0 = nt * 16;
  int e0 = blockIdx.x * 32 + rowg * 16;
  int e  = e0 + m;
  int sN = eidx[e];
  int dN = eidx[N_EDGES + e];
  const u16* hd = hbf + (size_t)dN * 64;
  const u16* hs = hbf + (size_t)sN * 64;
  const u16* ep = ebf + (size_t)e * 64;
  if (e + 32 < N_EDGES)
    __builtin_prefetch(ebf + (size_t)(e + 32) * 64, 0, 0);  // global_prefetch_b8
  const u16* bases[6] = { hd, hd + 32, hs, hs + 32, ep, ep + 32 };
  __syncthreads();
  v8f c = {};
#pragma unroll
  for (int ch = 0; ch < 6; ++ch) {
    FragBF A, B;
    const u16* b = bases[ch];
    A.q[0] = *(const uint4*)(b + half * 8);
    A.q[1] = *(const uint4*)(b + 16 + half * 8);
    const u16* wp = wpack + ((size_t)((ch * 4 + nt) * 32 + lane)) * 16;
    B.q[0] = *(const uint4*)wp;
    B.q[1] = *(const uint4*)(wp + 8);
    c = __builtin_amdgcn_wmma_f32_16x16x32_bf16(false, A.v, false, B.v, (short)0, c, false, false);
  }
  float s = 0.f, ss = 0.f;
  tile_writeout<false>(stile[wave], c, lane, half, m, mbf, e0, n0, N_EDGES, s, ss);
  s  += __shfl_xor(s, 16, 32);
  ss += __shfl_xor(ss, 16, 32);
  int col = n0 + m;
  if (half == 0) { atomicAdd(&lsum[col], s); atomicAdd(&lss[col], ss); }
  __syncthreads();
  if (tid < 64) { atomicAdd(&gsum[tid], lsum[tid]); atomicAdd(&gss[tid], lss[tid]); }
}

// Fold BN: scale = g*rsqrt(var+eps); shift = be - scale*mean  (linear bias cancels)
__global__ void k_bn_finalize(float* __restrict__ stage, const float* __restrict__ g,
                              const float* __restrict__ be, float invCnt) {
  int t = threadIdx.x;  // 64 threads
  float mean = stage[t] * invCnt;
  float var  = stage[64 + t] * invCnt - mean * mean;
  float sc   = g[t] * rsqrtf(var + BN_EPS);
  stage[128 + t] = sc;
  stage[192 + t] = be[t] - sc * mean;
}

// Generic second linear: y = relu(scale*x+shift) @ W2[64,64], IN-PLACE on buf,
// with fused stats. Block reads its rows fully, barriers, then writes same rows.
// GUARD=false for the edge path (E % 32 == 0): zero guard code in hot kernel.
template <bool GUARD>
__global__ void k_gemm2(u16* __restrict__ buf, const u16* __restrict__ wpack,
                        const float* __restrict__ scale, const float* __restrict__ shift,
                        float* __restrict__ gsum, float* __restrict__ gss, int rows) {
  __shared__ float lsum[64], lss[64];
  __shared__ float stile[8][16][17];
  int tid = threadIdx.x;
  if (tid < 64) { lsum[tid] = 0.f; lss[tid] = 0.f; }
  int lane = tid & 31, wave = tid >> 5, half = lane >> 4, m = lane & 15;
  int nt = wave & 3, rowg = wave >> 2, n0 = nt * 16;
  int row0 = blockIdx.x * 32 + rowg * 16;
  int ar = row0 + m;
  FragBF A[2];
  if (!GUARD || ar < rows) {
    const u16* rp = buf + (size_t)ar * 64;
#pragma unroll
    for (int ch = 0; ch < 2; ++ch) {
      FragBF tf;
      tf.q[0] = *(const uint4*)(rp + ch * 32 + half * 8);
      tf.q[1] = *(const uint4*)(rp + ch * 32 + 16 + half * 8);
#pragma unroll
      for (int j = 0; j < 16; ++j) {
        int kk = ch * 32 + ((j < 8) ? (half * 8 + j) : (16 + half * 8 + (j - 8)));
        float xv = bf2f(tf.s[j]);
        float y  = fmaxf(fmaf(scale[kk], xv, shift[kk]), 0.f);
        A[ch].s[j] = f2bf(y);
      }
    }
  } else {
    A[0].q[0] = make_uint4(0, 0, 0, 0); A[0].q[1] = make_uint4(0, 0, 0, 0);
    A[1].q[0] = make_uint4(0, 0, 0, 0); A[1].q[1] = make_uint4(0, 0, 0, 0);
  }
  __syncthreads();  // all in-place reads complete before writes below
  v8f c = {};
#pragma unroll
  for (int ch = 0; ch < 2; ++ch) {
    FragBF B;
    const u16* wp = wpack + ((size_t)((ch * 4 + nt) * 32 + lane)) * 16;
    B.q[0] = *(const uint4*)wp;
    B.q[1] = *(const uint4*)(wp + 8);
    c = __builtin_amdgcn_wmma_f32_16x16x32_bf16(false, A[ch].v, false, B.v, (short)0, c, false, false);
  }
  float s = 0.f, ss = 0.f;
  tile_writeout<GUARD>(stile[wave], c, lane, half, m, buf, row0, n0, rows, s, ss);
  s  += __shfl_xor(s, 16, 32);
  ss += __shfl_xor(ss, 16, 32);
  int col = n0 + m;
  if (half == 0) { atomicAdd(&lsum[col], s); atomicAdd(&lss[col], ss); }
  __syncthreads();
  if (tid < 64) { atomicAdd(&gsum[tid], lsum[tid]); atomicAdd(&gss[tid], lss[tid]); }
}

// edge finalize: m = relu(bn2(m2)); e += m (f32 + bf16 copy); atomic scatter agg[dst]+=m
__global__ void k_edge_fin(const u16* __restrict__ mbf, const float* __restrict__ scale,
                           const float* __restrict__ shift, const int* __restrict__ eidx,
                           float* __restrict__ e32, u16* __restrict__ ebf,
                           float* __restrict__ agg) {
  int t = blockIdx.x * 256 + threadIdx.x;
  if (t >= N_EDGES * 16) return;
  int e = t >> 4, c0 = (t & 15) * 4;
  int d = eidx[N_EDGES + e];
  const u16* mp = mbf + (size_t)e * 64 + c0;
  float m0 = fmaxf(fmaf(scale[c0 + 0], bf2f(mp[0]), shift[c0 + 0]), 0.f);
  float m1 = fmaxf(fmaf(scale[c0 + 1], bf2f(mp[1]), shift[c0 + 1]), 0.f);
  float m2 = fmaxf(fmaf(scale[c0 + 2], bf2f(mp[2]), shift[c0 + 2]), 0.f);
  float m3 = fmaxf(fmaf(scale[c0 + 3], bf2f(mp[3]), shift[c0 + 3]), 0.f);
  float4 ev = *(const float4*)(e32 + (size_t)e * 64 + c0);
  ev.x += m0; ev.y += m1; ev.z += m2; ev.w += m3;
  *(float4*)(e32 + (size_t)e * 64 + c0) = ev;
  uint2 pk; pk.x = pack2bf(ev.x, ev.y); pk.y = pack2bf(ev.z, ev.w);
  *(uint2*)(ebf + (size_t)e * 64 + c0) = pk;
  float* ap = agg + (size_t)d * 64 + c0;  // N-node rows: L2-resident atomics
  atomicAdd(ap + 0, m0); atomicAdd(ap + 1, m1);
  atomicAdd(ap + 2, m2); atomicAdd(ap + 3, m3);
}

// upd GEMM1: concat(h, agg) [N,128] @ W1[128,64] -> ubf pre-BN + stats
__global__ void k_upd_gemm1(const u16* __restrict__ hbf, const float* __restrict__ agg,
                            const u16* __restrict__ wpack, u16* __restrict__ ubf,
                            float* __restrict__ gsum, float* __restrict__ gss) {
  __shared__ float lsum[64], lss[64];
  __shared__ float stile[8][16][17];
  int tid = threadIdx.x;
  if (tid < 64) { lsum[tid] = 0.f; lss[tid] = 0.f; }
  int lane = tid & 31, wave = tid >> 5, half = lane >> 4, m = lane & 15;
  int nt = wave & 3, rowg = wave >> 2, n0 = nt * 16;
  int row0 = blockIdx.x * 32 + rowg * 16;
  int ar = row0 + m;
  FragBF A[4];
  if (ar < N_NODES) {
    const u16* hp = hbf + (size_t)ar * 64;
#pragma unroll
    for (int ch = 0; ch < 2; ++ch) {
      A[ch].q[0] = *(const uint4*)(hp + ch * 32 + half * 8);
      A[ch].q[1] = *(const uint4*)(hp + ch * 32 + 16 + half * 8);
    }
    const float* ap = agg + (size_t)ar * 64;
#pragma unroll
    for (int ch = 2; ch < 4; ++ch) {
      int base = (ch - 2) * 32;
#pragma unroll
      for (int j = 0; j < 8; ++j) A[ch].s[j]     = f2bf(ap[base + half * 8 + j]);
#pragma unroll
      for (int j = 0; j < 8; ++j) A[ch].s[8 + j] = f2bf(ap[base + 16 + half * 8 + j]);
    }
  } else {
#pragma unroll
    for (int ch = 0; ch < 4; ++ch) {
      A[ch].q[0] = make_uint4(0, 0, 0, 0);
      A[ch].q[1] = make_uint4(0, 0, 0, 0);
    }
  }
  __syncthreads();
  v8f c = {};
#pragma unroll
  for (int ch = 0; ch < 4; ++ch) {
    FragBF B;
    const u16* wp = wpack + ((size_t)((ch * 4 + nt) * 32 + lane)) * 16;
    B.q[0] = *(const uint4*)wp;
    B.q[1] = *(const uint4*)(wp + 8);
    c = __builtin_amdgcn_wmma_f32_16x16x32_bf16(false, A[ch].v, false, B.v, (short)0, c, false, false);
  }
  float s = 0.f, ss = 0.f;
  tile_writeout<true>(stile[wave], c, lane, half, m, ubf, row0, n0, N_NODES, s, ss);
  s  += __shfl_xor(s, 16, 32);
  ss += __shfl_xor(ss, 16, 32);
  int col = n0 + m;
  if (half == 0) { atomicAdd(&lsum[col], s); atomicAdd(&lss[col], ss); }
  __syncthreads();
  if (tid < 64) { atomicAdd(&gsum[tid], lsum[tid]); atomicAdd(&gss[tid], lss[tid]); }
}

// node update: h += relu(bn2(u2)); refresh bf16 copy
__global__ void k_node_upd(const u16* __restrict__ ubf, const float* __restrict__ scale,
                           const float* __restrict__ shift,
                           float* __restrict__ h, u16* __restrict__ hbf) {
  int t = blockIdx.x * 256 + threadIdx.x;
  if (t >= N_NODES * 16) return;
  int n = t >> 4, c0 = (t & 15) * 4;
  const u16* up = ubf + (size_t)n * 64 + c0;
  float u0 = fmaxf(fmaf(scale[c0 + 0], bf2f(up[0]), shift[c0 + 0]), 0.f);
  float u1 = fmaxf(fmaf(scale[c0 + 1], bf2f(up[1]), shift[c0 + 1]), 0.f);
  float u2 = fmaxf(fmaf(scale[c0 + 2], bf2f(up[2]), shift[c0 + 2]), 0.f);
  float u3 = fmaxf(fmaf(scale[c0 + 3], bf2f(up[3]), shift[c0 + 3]), 0.f);
  float4 hv = *(const float4*)(h + (size_t)n * 64 + c0);
  hv.x += u0; hv.y += u1; hv.z += u2; hv.w += u3;
  *(float4*)(h + (size_t)n * 64 + c0) = hv;
  uint2 pk; pk.x = pack2bf(hv.x, hv.y); pk.y = pack2bf(hv.z, hv.w);
  *(uint2*)(hbf + (size_t)n * 64 + c0) = pk;
}

// out = e @ pred_w + pred_b : per-edge dot over 64 cols (HBM-bound stream)
__global__ void k_pred(const float* __restrict__ e32, const float* __restrict__ pw,
                       const float* __restrict__ pb, float* __restrict__ out) {
  int e = blockIdx.x * 256 + threadIdx.x;
  if (e >= N_EDGES) return;
  const float* row = e32 + (size_t)e * 64;
  float acc = pb[0];
#pragma unroll
  for (int j = 0; j < 64; j += 4) {
    float4 v = *(const float4*)(row + j);
    acc = fmaf(v.x, pw[j + 0], acc);
    acc = fmaf(v.y, pw[j + 1], acc);
    acc = fmaf(v.z, pw[j + 2], acc);
    acc = fmaf(v.w, pw[j + 3], acc);
  }
  out[e] = acc;
}

// ---------------------------------------------------------------------------
extern "C" void kernel_launch(void* const* d_in, const int* in_sizes, int n_in,
                              void* d_out, int out_size, void* d_ws, size_t ws_size,
                              hipStream_t stream) {
  const float* x       = (const float*)d_in[0];
  const int*   eidx    = (const int*)  d_in[1];
  const float* eattr   = (const float*)d_in[2];
  const float* lin_w   = (const float*)d_in[3];
  const float* lin_b   = (const float*)d_in[4];
  const float* edg_w   = (const float*)d_in[5];
  const float* edg_b   = (const float*)d_in[6];
  const float* msg_w1  = (const float*)d_in[7];
  const float* msg_g1  = (const float*)d_in[9];
  const float* msg_be1 = (const float*)d_in[10];
  const float* msg_w2  = (const float*)d_in[11];
  const float* msg_g2  = (const float*)d_in[13];
  const float* msg_be2 = (const float*)d_in[14];
  const float* upd_w1  = (const float*)d_in[15];
  const float* upd_g1  = (const float*)d_in[17];
  const float* upd_be1 = (const float*)d_in[18];
  const float* upd_w2  = (const float*)d_in[19];
  const float* upd_g2  = (const float*)d_in[21];
  const float* upd_be2 = (const float*)d_in[22];
  const float* pred_w  = (const float*)d_in[23];
  const float* pred_b  = (const float*)d_in[24];
  float* out = (float*)d_out;

  char* ws = (char*)d_ws;
  u16*   w_lin  = (u16*)  (ws + OFF_WLIN);
  u16*   w_msg1 = (u16*)  (ws + OFF_WMSG1);
  u16*   w_msg2 = (u16*)  (ws + OFF_WMSG2);
  u16*   w_upd1 = (u16*)  (ws + OFF_WUPD1);
  u16*   w_upd2 = (u16*)  (ws + OFF_WUPD2);
  float* stats  = (float*)(ws + OFF_STATS);
  float* h      = (float*)(ws + OFF_H);
  u16*   hbf    = (u16*)  (ws + OFF_HBF);
  float* agg    = (float*)(ws + OFF_AGG);
  u16*   ubf    = (u16*)  (ws + OFF_UBF);
  float* e32    = (float*)(ws + OFF_E);
  u16*   ebf    = (u16*)  (ws + OFF_EBF);
  u16*   mbf    = (u16*)  (ws + OFF_MBF);

  // Zero all BN stat accumulators (12 stages x 256 f32) every call (graph-safe).
  k_zero<<<12, 256, 0, stream>>>(stats, 12 * 256);

  // Pack weights into per-lane WMMA B-fragment layout (bf16).
  k_pack_w<<<8, 256, 0, stream>>>(lin_w, w_lin, 32);
  for (int l = 0; l < NLAYERS; ++l) {
    k_pack_w<<<48, 256, 0, stream>>>(msg_w1 + (size_t)l * 192 * 64, w_msg1 + (size_t)l * 6 * 2048, 192);
    k_pack_w<<<16, 256, 0, stream>>>(msg_w2 + (size_t)l * 64 * 64,  w_msg2 + (size_t)l * 2 * 2048, 64);
    k_pack_w<<<32, 256, 0, stream>>>(upd_w1 + (size_t)l * 128 * 64, w_upd1 + (size_t)l * 4 * 2048, 128);
    k_pack_w<<<16, 256, 0, stream>>>(upd_w2 + (size_t)l * 64 * 64,  w_upd2 + (size_t)l * 2 * 2048, 64);
  }

  // Input projections.
  k_h_init<<<(N_NODES + 31) / 32, 256, 0, stream>>>(x, w_lin, lin_b, h, hbf);
  k_e_init<<<(N_EDGES * 16) / 256, 256, 0, stream>>>(eattr, edg_w, edg_b, e32, ebf);

  const float invE = 1.f / (float)N_EDGES;
  const float invN = 1.f / (float)N_NODES;

  for (int l = 0; l < NLAYERS; ++l) {
    float* st_m1 = stats + (l * 4 + 0) * 256;
    float* st_m2 = stats + (l * 4 + 1) * 256;
    float* st_u1 = stats + (l * 4 + 2) * 256;
    float* st_u2 = stats + (l * 4 + 3) * 256;

    // message MLP over edges (hot path: GUARD=false, E % 32 == 0)
    k_msg_gemm1<<<N_EDGES / 32, 256, 0, stream>>>(eidx, hbf, ebf,
        w_msg1 + (size_t)l * 6 * 2048, mbf, st_m1, st_m1 + 64);
    k_bn_finalize<<<1, 64, 0, stream>>>(st_m1, msg_g1 + l * 64, msg_be1 + l * 64, invE);
    k_gemm2<false><<<N_EDGES / 32, 256, 0, stream>>>(mbf, w_msg2 + (size_t)l * 2 * 2048,
        st_m1 + 128, st_m1 + 192, st_m2, st_m2 + 64, N_EDGES);
    k_bn_finalize<<<1, 64, 0, stream>>>(st_m2, msg_g2 + l * 64, msg_be2 + l * 64, invE);

    // aggregate + edge residual
    k_zero<<<(N_NODES * 64 + 255) / 256, 256, 0, stream>>>(agg, N_NODES * 64);
    k_edge_fin<<<(N_EDGES * 16) / 256, 256, 0, stream>>>(mbf, st_m2 + 128, st_m2 + 192,
        eidx, e32, ebf, agg);

    // update MLP over nodes (cold path: guarded)
    k_upd_gemm1<<<(N_NODES + 31) / 32, 256, 0, stream>>>(hbf, agg,
        w_upd1 + (size_t)l * 4 * 2048, ubf, st_u1, st_u1 + 64);
    k_bn_finalize<<<1, 64, 0, stream>>>(st_u1, upd_g1 + l * 64, upd_be1 + l * 64, invN);
    k_gemm2<true><<<(N_NODES + 31) / 32, 256, 0, stream>>>(ubf, w_upd2 + (size_t)l * 2 * 2048,
        st_u1 + 128, st_u1 + 192, st_u2, st_u2 + 64, N_NODES);
    k_bn_finalize<<<1, 64, 0, stream>>>(st_u2, upd_g2 + l * 64, upd_be2 + l * 64, invN);
    k_node_upd<<<(N_NODES * 16 + 255) / 256, 256, 0, stream>>>(ubf, st_u2 + 128, st_u2 + 192, h, hbf);
  }

  // prediction head
  k_pred<<<(N_EDGES + 255) / 256, 256, 0, stream>>>(e32, pred_w, pred_b, out);
}
